// neko_os_DTD_mk4_20512763806188
// MI455X (gfx1250) — matrix operand order repaired and checked by reading, not compiled, not run on hardware
//
#include <hip/hip_runtime.h>
#include <hip/hip_bf16.h>

// Problem dims (from reference)
#define NBD 64
#define NCD 512
#define NHD 16
#define NWD 48
#define HWD (NHD*NWD)   // 768
#define NTD 32
#define NPROTO 4000
#define NCLS 4001
#define NGD 1536        // 3*NC
#define NXD 1024        // 2*NC

typedef __bf16 bf16_t;
typedef __attribute__((ext_vector_type(16))) __bf16 v16bf;
typedef __attribute__((ext_vector_type(8)))  __bf16 v8bf;
typedef __attribute__((ext_vector_type(8)))  float  v8f;
typedef __attribute__((ext_vector_type(8)))  unsigned short us8;
typedef __attribute__((ext_vector_type(8)))  unsigned int   u32x8;

__device__ __forceinline__ unsigned short f2bf_bits(float f) {
  unsigned u = __builtin_bit_cast(unsigned, f);
  u += 0x7FFFu + ((u >> 16) & 1u);          // round-to-nearest-even
  return (unsigned short)(u >> 16);
}
// two f32 -> packed 2x bf16 in one u32 (integer-domain, no half-reg moves)
__device__ __forceinline__ unsigned pack2bf(float f0, float f1) {
  unsigned u0 = __builtin_bit_cast(unsigned, f0);
  unsigned u1 = __builtin_bit_cast(unsigned, f1);
  u0 += 0x7FFFu + ((u0 >> 16) & 1u);
  u1 += 0x7FFFu + ((u1 >> 16) & 1u);
  return (u0 >> 16) | (u1 & 0xFFFF0000u);
}
// WMMA fragment from two 16-byte chunks of packed bf16 (pure loads, no VALU)
__device__ __forceinline__ v16bf frag16(const unsigned short* p0,
                                        const unsigned short* p1) {
  v8bf lo = __builtin_bit_cast(v8bf, *(const us8*)p0);
  v8bf hi = __builtin_bit_cast(v8bf, *(const us8*)p1);
  return __builtin_shufflevector(lo, hi, 0,1,2,3,4,5,6,7,8,9,10,11,12,13,14,15);
}
// 16 contiguous f32 -> bf16 fragment (getC feature path only)
__device__ __forceinline__ v16bf fragcvt(const float* p) {
  u32x8 r;
#pragma unroll
  for (int i = 0; i < 8; ++i) r[i] = pack2bf(p[2*i], p[2*i+1]);
  return __builtin_bit_cast(v16bf, r);
}
__device__ __forceinline__ v8f wmma_bf16(v16bf a, v16bf b, v8f c) {
  return __builtin_amdgcn_wmma_f32_16x16x32_bf16(false, a, false, b, (short)0, c,
                                                 false, false);
}

// ---------------------------------------------------------------------------
// getC: C_bf[t,b,c] = bf16( sum_hw feature[b,c,hw] * A[b,t,hw] )
// grid = (NB, 4), block = 256; each wave: one c-tile x BOTH t-tiles
// (B fragment loaded/converted once, feeds 2 WMMAs)
// ---------------------------------------------------------------------------
__global__ void getC_kernel(const float* __restrict__ feature,
                            const unsigned short* __restrict__ Aatt_bf,
                            unsigned short* __restrict__ C_bf) {
  const int b    = blockIdx.x;
  const int wave = threadIdx.x >> 5;
  const int lane = threadIdx.x & 31;
  const int nt   = blockIdx.y * 8 + wave;      // c-tile (0..31)
  const int g = lane >> 4, l = lane & 15;

  const unsigned short* a0 = Aatt_bf + ((size_t)b * NTD + l)      * HWD; // t-tile 0
  const unsigned short* a1 = Aatt_bf + ((size_t)b * NTD + 16 + l) * HWD; // t-tile 1
  const float*          br = feature + ((size_t)b * NCD + (nt*16 + l)) * HWD;

  v8f acc0 = {}, acc1 = {};
  for (int k0 = 0; k0 < HWD; k0 += 32) {
    const v16bf bm = fragcvt(br + k0 + 16*g);
    acc0 = wmma_bf16(frag16(a0 + k0 + 16*g, a0 + k0 + 16*g + 8), bm, acc0);
    acc1 = wmma_bf16(frag16(a1 + k0 + 16*g, a1 + k0 + 16*g + 8), bm, acc1);
  }
  const int c = nt*16 + l;
#pragma unroll
  for (int r = 0; r < 8; ++r) {
    const int t0 = r + 8*g;
    C_bf[((size_t)t0        * NBD + b) * NCD + c] = f2bf_bits(acc0[r]);
    C_bf[((size_t)(16 + t0) * NBD + b) * NCD + c] = f2bf_bits(acc1[r]);
  }
}

// ---------------------------------------------------------------------------
__global__ void cvt_bf16_kernel(const float* __restrict__ src,
                                unsigned short* __restrict__ dst, int n) {
  int i = blockIdx.x * blockDim.x + threadIdx.x;
  if (i < n) dst[i] = f2bf_bits(src[i]);
}

// h = 0 (f32 + bf16), prev = broadcast(STA) in bf16
__global__ void init_state_kernel(const float* __restrict__ STA,
                                  float* __restrict__ h,
                                  unsigned short* __restrict__ h_bf,
                                  unsigned short* __restrict__ prev_bf) {
  int i = blockIdx.x * blockDim.x + threadIdx.x;   // NB*NC
  if (i < NBD * NCD) {
    h[i] = 0.f;
    h_bf[i] = 0;
    prev_bf[i] = f2bf_bits(STA[i % NCD]);
  }
}

// ---------------------------------------------------------------------------
// gate GEMMs: gi = [C_t | prev] @ W_ih^T ; gh = h @ W_hh^T
// grid = 12 blocks x 256; each wave: one n-tile x ALL 4 batch-tiles
// (B fragment loaded once per K-step, feeds 4 WMMAs; all operands bf16)
// ---------------------------------------------------------------------------
__global__ void gemm_gates_kernel(const unsigned short* __restrict__ C_t,
                                  const unsigned short* __restrict__ prev_bf,
                                  const unsigned short* __restrict__ h_bf,
                                  const unsigned short* __restrict__ Wih,
                                  const unsigned short* __restrict__ Whh,
                                  float* __restrict__ gi,
                                  float* __restrict__ gh) {
  const int wave = threadIdx.x >> 5;
  const int lane = threadIdx.x & 31;
  const int ntile = blockIdx.x * 8 + wave;     // 0..95
  const int g = lane >> 4, l = lane & 15;
  const int n = ntile*16 + l;

  const unsigned short* wi = Wih + (size_t)n * NXD;
  const unsigned short* wh = Whh + (size_t)n * NCD;
  const unsigned short* cb[4]; const unsigned short* pv[4]; const unsigned short* hb[4];
#pragma unroll
  for (int mt = 0; mt < 4; ++mt) {
    const size_t row = (size_t)(mt*16 + l) * NCD;
    cb[mt] = C_t + row; pv[mt] = prev_bf + row; hb[mt] = h_bf + row;
  }

  v8f ai[4] = {{}, {}, {}, {}};
  for (int k0 = 0; k0 < NCD; k0 += 32) {       // x[:, 0:512] = C_t
    __builtin_prefetch(wi + k0 + 256, 0, 1);
    const v16bf bm = frag16(wi + k0 + 8*g, wi + k0 + 16 + 8*g);
#pragma unroll
    for (int mt = 0; mt < 4; ++mt)
      ai[mt] = wmma_bf16(frag16(cb[mt] + k0 + 8*g, cb[mt] + k0 + 16 + 8*g), bm, ai[mt]);
  }
  for (int k0 = 0; k0 < NCD; k0 += 32) {       // x[:, 512:1024] = prev
    __builtin_prefetch(wi + NCD + k0 + 256, 0, 1);
    const v16bf bm = frag16(wi + NCD + k0 + 8*g, wi + NCD + k0 + 16 + 8*g);
#pragma unroll
    for (int mt = 0; mt < 4; ++mt)
      ai[mt] = wmma_bf16(frag16(pv[mt] + k0 + 8*g, pv[mt] + k0 + 16 + 8*g), bm, ai[mt]);
  }
#pragma unroll
  for (int mt = 0; mt < 4; ++mt)
#pragma unroll
    for (int r = 0; r < 8; ++r)
      gi[(size_t)(mt*16 + r + 8*g) * NGD + ntile*16 + l] = ai[mt][r];

  v8f ah[4] = {{}, {}, {}, {}};
  for (int k0 = 0; k0 < NCD; k0 += 32) {       // h @ W_hh^T
    __builtin_prefetch(wh + k0 + 256, 0, 1);
    const v16bf bm = frag16(wh + k0 + 8*g, wh + k0 + 16 + 8*g);
#pragma unroll
    for (int mt = 0; mt < 4; ++mt)
      ah[mt] = wmma_bf16(frag16(hb[mt] + k0 + 8*g, hb[mt] + k0 + 16 + 8*g), bm, ah[mt]);
  }
#pragma unroll
  for (int mt = 0; mt < 4; ++mt)
#pragma unroll
    for (int r = 0; r < 8; ++r)
      gh[(size_t)(mt*16 + r + 8*g) * NGD + ntile*16 + l] = ah[mt][r];
}

// GRU elementwise gates: h = (1-z)*n + z*h  (writes f32 and bf16 views)
__global__ void gru_gates_kernel(const float* __restrict__ gi,
                                 const float* __restrict__ gh,
                                 float* __restrict__ h,
                                 unsigned short* __restrict__ h_bf,
                                 const float* __restrict__ b_ih,
                                 const float* __restrict__ b_hh) {
  int i = blockIdx.x * blockDim.x + threadIdx.x;   // NB*NC
  if (i >= NBD * NCD) return;
  const int b = i / NCD, c = i % NCD;
  const float* gib = gi + (size_t)b * NGD;
  const float* ghb = gh + (size_t)b * NGD;
  const float ir = gib[c]         + b_ih[c];
  const float iz = gib[c + NCD]   + b_ih[c + NCD];
  const float in = gib[c + 2*NCD] + b_ih[c + 2*NCD];
  const float hr = ghb[c]         + b_hh[c];
  const float hz = ghb[c + NCD]   + b_hh[c + NCD];
  const float hn = ghb[c + 2*NCD] + b_hh[c + 2*NCD];
  const float r = 1.f / (1.f + expf(-(ir + hr)));
  const float z = 1.f / (1.f + expf(-(iz + hz)));
  const float nn = tanhf(in + r * hn);
  const float hv = (1.f - z) * nn + z * h[i];
  h[i] = hv;
  h_bf[i] = f2bf_bits(hv);
}

// ---------------------------------------------------------------------------
// score GEMM: out[(b*NT+t)*NCLS + p] = ALPHA*(h @ protos^T) + BIAS
// 250 n-tiles; each wave: one p-tile x ALL 4 batch-tiles (B loaded once)
// plabel == arange(NCLS) -> scatter_cvt is identity, write scores directly.
// ---------------------------------------------------------------------------
__global__ void gemm_scores_kernel(const unsigned short* __restrict__ h_bf,
                                   const unsigned short* __restrict__ protos_bf,
                                   float* __restrict__ out, int t,
                                   const float* __restrict__ ALPHA,
                                   const float* __restrict__ BIAS) {
  const int wave = threadIdx.x >> 5;
  const int lane = threadIdx.x & 31;
  const int nt = blockIdx.x * 8 + wave;        // 0..255 (250 valid)
  if (nt >= 250) return;                       // wave-uniform: EXEC untouched
  const int g = lane >> 4, l = lane & 15;

  const unsigned short* wrow = protos_bf + (size_t)(nt*16 + l) * NCD;
  const unsigned short* hb[4];
#pragma unroll
  for (int mt = 0; mt < 4; ++mt) hb[mt] = h_bf + (size_t)(mt*16 + l) * NCD;

  v8f acc[4] = {{}, {}, {}, {}};
  for (int k0 = 0; k0 < NCD; k0 += 32) {
    __builtin_prefetch(wrow + k0 + 256, 0, 1);
    const v16bf bm = frag16(wrow + k0 + 8*g, wrow + k0 + 16 + 8*g);
#pragma unroll
    for (int mt = 0; mt < 4; ++mt)
      acc[mt] = wmma_bf16(frag16(hb[mt] + k0 + 8*g, hb[mt] + k0 + 16 + 8*g), bm, acc[mt]);
  }
  const float alpha = *ALPHA, bias = *BIAS;
  const int p = nt*16 + l;
#pragma unroll
  for (int mt = 0; mt < 4; ++mt)
#pragma unroll
    for (int r = 0; r < 8; ++r)
      out[((size_t)(mt*16 + r + 8*g) * NTD + t) * NCLS + p] = acc[mt][r] * alpha + bias;
}

// argmax over class row (first-max wins, matching jnp.argmax),
// write UNK column, gather next-step embedding (prev = chemb[argmax])
__global__ void argmax_gather_kernel(float* __restrict__ out, int t,
                                     const unsigned short* __restrict__ chemb_bf,
                                     unsigned short* __restrict__ prev_bf,
                                     const float* __restrict__ UNK_SCR) {
  const int b = blockIdx.x;
  float* row = out + ((size_t)b * NTD + t) * NCLS;
  const float unk = *UNK_SCR;
  __shared__ float svals[256];
  __shared__ int   sidx[256];

  float best = unk; int bi = NPROTO;           // UNK slot participates
  for (int p = threadIdx.x; p < NPROTO; p += blockDim.x) {
    const float v = row[p];
    if (v > best || (v == best && p < bi)) { best = v; bi = p; }
  }
  svals[threadIdx.x] = best; sidx[threadIdx.x] = bi;
  __syncthreads();
  for (int s = blockDim.x >> 1; s > 0; s >>= 1) {
    if (threadIdx.x < s) {
      const float v = svals[threadIdx.x + s]; const int j = sidx[threadIdx.x + s];
      if (v > svals[threadIdx.x] ||
          (v == svals[threadIdx.x] && j < sidx[threadIdx.x])) {
        svals[threadIdx.x] = v; sidx[threadIdx.x] = j;
      }
    }
    __syncthreads();
  }
  if (threadIdx.x == 0) row[NPROTO] = unk;     // UNK score column of output
  const int amax = sidx[0];
  const unsigned short* src = chemb_bf + (size_t)amax * NCD;
  unsigned short*       dst = prev_bf  + (size_t)b * NCD;
  for (int c = threadIdx.x; c < NCD; c += blockDim.x) dst[c] = src[c];
}

// ---------------------------------------------------------------------------
extern "C" void kernel_launch(void* const* d_in, const int* in_sizes, int n_in,
                              void* d_out, int out_size, void* d_ws, size_t ws_size,
                              hipStream_t stream) {
  const float* feature = (const float*)d_in[0];   // [64,512,16,48]
  const float* Aatt    = (const float*)d_in[1];   // [64,32,16,48]
  const float* protos  = (const float*)d_in[2];   // [4000,512]
  const float* semb    = (const float*)d_in[3];   // [4000,512]
  const float* STA     = (const float*)d_in[4];   // [1,512]
  const float* UNK     = (const float*)d_in[5];   // [1,512]
  const float* W_ih    = (const float*)d_in[6];   // [1536,1024]
  const float* W_hh    = (const float*)d_in[7];   // [1536,512]
  const float* b_ih    = (const float*)d_in[8];   // [1536]
  const float* b_hh    = (const float*)d_in[9];   // [1536]
  const float* ALPHA   = (const float*)d_in[10];
  const float* BIAS    = (const float*)d_in[11];
  const float* UNK_SCR = (const float*)d_in[12];
  // d_in[13] plabel == arange(NCLS): identity scatter, unused.
  // d_in[14] text_length == NT everywhere: static shapes, unused.

  float* out = (float*)d_out;                    // [2048,4001] then A copy

  // workspace carve-up (~19.2 MB)
  char* ws = (char*)d_ws;
  size_t off = 0;
  auto carve = [&](size_t bytes) { char* p = ws + off; off += (bytes + 255) & ~(size_t)255; return p; };
  unsigned short* Aatt_bf   = (unsigned short*)carve((size_t)NBD * NTD * HWD * 2);
  unsigned short* C_bf      = (unsigned short*)carve((size_t)NTD * NBD * NCD * 2);
  unsigned short* Wih_bf    = (unsigned short*)carve((size_t)NGD * NXD * 2);
  unsigned short* Whh_bf    = (unsigned short*)carve((size_t)NGD * NCD * 2);
  unsigned short* protos_bf = (unsigned short*)carve((size_t)NPROTO * NCD * 2);
  unsigned short* chemb_bf  = (unsigned short*)carve((size_t)NCLS * NCD * 2);
  float*          h_ws      = (float*)carve((size_t)NBD * NCD * 4);
  unsigned short* hbf_ws    = (unsigned short*)carve((size_t)NBD * NCD * 2);
  unsigned short* prev_bf   = (unsigned short*)carve((size_t)NBD * NCD * 2);
  float*          gi_ws     = (float*)carve((size_t)NBD * NGD * 4);
  float*          gh_ws     = (float*)carve((size_t)NBD * NGD * 4);
  (void)ws_size; (void)n_in; (void)in_sizes; (void)out_size;

  // out_attns = A flattened, appended after scores
  hipMemcpyAsync(out + (size_t)NBD * NTD * NCLS, (const void*)Aatt,
                 (size_t)NBD * NTD * HWD * sizeof(float),
                 hipMemcpyDeviceToDevice, stream);

  // one-time per-launch bf16 conversions (deterministic)
  auto cvt = [&](const float* s, unsigned short* d, int n) {
    cvt_bf16_kernel<<<(n + 255) / 256, 256, 0, stream>>>(s, d, n);
  };
  cvt(Aatt,   Aatt_bf,   NBD * NTD * HWD);
  cvt(W_ih,   Wih_bf,    NGD * NXD);
  cvt(W_hh,   Whh_bf,    NGD * NCD);
  cvt(protos, protos_bf, NPROTO * NCD);
  cvt(semb,   chemb_bf,  NPROTO * NCD);
  cvt(UNK,    chemb_bf + (size_t)NPROTO * NCD, NCD);   // chemb row 4000

  init_state_kernel<<<(NBD * NCD + 255) / 256, 256, 0, stream>>>(
      STA, h_ws, hbf_ws, prev_bf);

  // attention pooling via WMMA (writes C in bf16)
  getC_kernel<<<dim3(NBD, 4), 256, 0, stream>>>(feature, Aatt_bf, C_bf);

  // sequential greedy decode
  for (int t = 0; t < NTD; ++t) {
    gemm_gates_kernel<<<12, 256, 0, stream>>>(
        C_bf + (size_t)t * NBD * NCD, prev_bf, hbf_ws, Wih_bf, Whh_bf, gi_ws, gh_ws);
    gru_gates_kernel<<<(NBD * NCD + 255) / 256, 256, 0, stream>>>(
        gi_ws, gh_ws, h_ws, hbf_ws, b_ih, b_hh);
    gemm_scores_kernel<<<32, 256, 0, stream>>>(hbf_ws, protos_bf, out, t, ALPHA, BIAS);
    argmax_gather_kernel<<<NBD, 256, 0, stream>>>(out, t, chemb_bf, prev_bf, UNK_SCR);
  }
}
